// SNNGCN_87677462380802
// MI455X (gfx1250) — compile-verified, matrix operand order checked by reading, not run on hardware
//
#include <hip/hip_runtime.h>
#include <hip/hip_bf16.h>

// ---------------------------------------------------------------------------
// Spiking GCN on MI455X (gfx1250, wave32, WMMA + TDM).
//  - GEMMs: f16 WMMA (binary spikes exact in f16), f32 accumulate.
//  - GEMM2 A-tile staged to LDS via Tensor Data Mover (tensor_load_to_lds,
//    s_wait_tensorcnt), descriptor built per CDNA5 D# bitfields.
//  - Edge scatter: one wave32 per edge, SGPR indices via readfirstlane,
//    hardware global_atomic_add_f32 (relaxed, agent scope).
//  - Algebraic folding: mean-div in LIF, self-loops in pool, b2/blin/cnt/T in
//    the single final readout.
// ---------------------------------------------------------------------------

#define N_NODES 50000
#define N_EDGES 800000
#define F_INP   128
#define HDIM    256
#define OUTC    10
#define NGRAPH  128
#define TSTEPS  15
#define INV_TAU 0.5f
#define THRV    0.25f

typedef __attribute__((ext_vector_type(16))) _Float16 v16h;
typedef __attribute__((ext_vector_type(8)))  _Float16 v8h;
typedef __attribute__((ext_vector_type(4)))  _Float16 h4;
typedef __attribute__((ext_vector_type(8)))  float    v8f;
typedef unsigned int u32x4 __attribute__((ext_vector_type(4)));
typedef int          i32x4 __attribute__((ext_vector_type(4)));
typedef int          i32x8 __attribute__((ext_vector_type(8)));

#if defined(__has_builtin)
#if __has_builtin(__builtin_amdgcn_tensor_load_to_lds) && \
    __has_builtin(__builtin_amdgcn_s_wait_tensorcnt)
#define HAVE_TDM 1
#endif
#endif
#ifndef HAVE_TDM
#define HAVE_TDM 0
#endif

// ----------------------------- helpers -------------------------------------
__device__ __forceinline__ void atomAddF(float* p, float v) {
  (void)__hip_atomic_fetch_add(p, v, __ATOMIC_RELAXED, __HIP_MEMORY_SCOPE_AGENT);
}

__device__ __forceinline__ unsigned hash_u32(unsigned x) {
  x ^= x >> 16; x *= 0x7feb352dU;
  x ^= x >> 15; x *= 0x846ca68bU;
  x ^= x >> 16;
  return x;
}
__device__ __forceinline__ float hash_u01(unsigned idx, unsigned t) {
  unsigned h = hash_u32(idx ^ (t * 0x9E3779B9U + 0x85ebca6bU));
  return (float)(h >> 8) * 0x1p-24f;
}

// 16-bit A/B fragment: two contiguous 8-half (16B) chunks 16 K-slots apart.
__device__ __forceinline__ v16h load_frag(const _Float16* p0) {
  v8h lo = *(const v8h*)(p0);
  v8h hi = *(const v8h*)(p0 + 16);
  v16h r;
#pragma unroll
  for (int i = 0; i < 8; ++i) { r[i] = lo[i]; r[i + 8] = hi[i]; }
  return r;
}

// --------------------------------- zero ------------------------------------
__global__ __launch_bounds__(256) void zero4_kernel(float4* __restrict__ p,
                                                    long long n4) {
  long long i = (long long)blockIdx.x * blockDim.x + threadIdx.x;
  if (i < n4) p[i] = make_float4(0.f, 0.f, 0.f, 0.f);
}

// --------------------------- degree / count prep ---------------------------
__global__ __launch_bounds__(256) void degree_kernel(const int* __restrict__ dst,
                                                     float* __restrict__ degc) {
  int e = blockIdx.x * blockDim.x + threadIdx.x;
  if (e < N_EDGES) atomAddF(&degc[dst[e]], 1.0f);
}

__global__ __launch_bounds__(256) void graphcnt_kernel(const int* __restrict__ batch,
                                                       float* __restrict__ cnt) {
  int n = blockIdx.x * blockDim.x + threadIdx.x;
  if (n < N_NODES) atomAddF(&cnt[batch[n]], 1.0f);
}

// deg (mean-agg) = max(edgecount,1); isw = 1/sqrt(edgecount + 1 selfloop)
__global__ __launch_bounds__(256) void finalize_deg_kernel(float* __restrict__ degc,
                                                           float* __restrict__ isw) {
  int n = blockIdx.x * blockDim.x + threadIdx.x;
  if (n >= N_NODES) return;
  float c = degc[n];
  isw[n]  = rsqrtf(c + 1.0f);
  degc[n] = fmaxf(c, 1.0f);
}

__global__ __launch_bounds__(256) void edgew_kernel(const int* __restrict__ src,
                                                    const int* __restrict__ dst,
                                                    const float* __restrict__ isw,
                                                    float* __restrict__ ew) {
  int e = blockIdx.x * blockDim.x + threadIdx.x;
  if (e < N_EDGES) ew[e] = isw[src[e]] * isw[dst[e]];
}

// ------------------------ weight transpose + f16 cast ----------------------
__global__ __launch_bounds__(256) void convert_w1_kernel(const float* __restrict__ W1,
                                                         _Float16* __restrict__ W1T) {
  int i = blockIdx.x * blockDim.x + threadIdx.x;   // i = h*F_INP + k
  if (i >= F_INP * HDIM) return;
  int h = i / F_INP, k = i % F_INP;
  W1T[i] = (_Float16)W1[k * HDIM + h];
}
__global__ __launch_bounds__(256) void convert_w2_kernel(const float* __restrict__ W2,
                                                         _Float16* __restrict__ W2T) {
  int i = blockIdx.x * blockDim.x + threadIdx.x;   // i = n*HDIM + k
  if (i >= HDIM * HDIM) return;
  int n = i / HDIM, k = i % HDIM;
  W2T[i] = (_Float16)W2[k * HDIM + n];
}

// --------------------- GEMM1: Poisson spikes @ W1 (WMMA) -------------------
// Block: 16 rows (N/16 = 3125 blocks), 256 threads = 8 waves; spikes generated
// once into LDS, each wave computes two 16x16 tiles (8 waves * 2 * 16 = 256).
__global__ __launch_bounds__(256) void spike_gemm1_kernel(
    const float* __restrict__ x, const _Float16* __restrict__ W1T,
    float* __restrict__ H1, int t) {
  __shared__ _Float16 At[16 * F_INP];            // 4 KB spike tile (16 x 128)
  const int rowbase = blockIdx.x * 16;
  const int tid = threadIdx.x;

#pragma unroll
  for (int j = 0; j < 8; ++j) {                  // 256 thr * 8 = 2048 elems
    int idx = tid * 8 + j;                       // idx = r*128 + k
    int r = idx >> 7, k = idx & 127;
    unsigned gidx = (unsigned)((rowbase + r) * F_INP + k);
    float u = hash_u01(gidx, (unsigned)t);
    At[idx] = (u < x[gidx]) ? (_Float16)1.0f : (_Float16)0.0f;
  }
  __syncthreads();

  const int lane = tid & 31, wave = tid >> 5;
  const int m    = lane & 15;                    // A row / B col / C col
  const int ksel = (lane & 16) ? 8 : 0;          // K phase per half-wave
  const int moff = (lane & 16) ? 8 : 0;          // C row offset per half-wave

#pragma unroll
  for (int ct = 0; ct < 2; ++ct) {
    const int colbase = (wave * 2 + ct) * 16;
    v8f c = {};
#pragma unroll
    for (int kb = 0; kb < F_INP / 32; ++kb) {
      v16h a = load_frag(&At[m * F_INP + kb * 32 + ksel]);
      v16h b = load_frag(&W1T[(size_t)(colbase + m) * F_INP + kb * 32 + ksel]);
      c = __builtin_amdgcn_wmma_f32_16x16x32_f16(false, a, false, b,
                                                 (short)0, c, false, false);
    }
    const int n = colbase + m;
#pragma unroll
    for (int r = 0; r < 8; ++r)
      H1[(size_t)(rowbase + r + moff) * HDIM + n] = c[r];
  }
}

// ------------------ GEMM2: spikes @ W2 (WMMA, TDM-staged A) ----------------
// A-tile (16 x 256 f16, 8KB) pulled into LDS by the Tensor Data Mover from
// one wave; s_wait_tensorcnt + workgroup barrier publishes it to all 8 waves.
__global__ __launch_bounds__(256) void spike_gemm2_kernel(
    const _Float16* __restrict__ S, const _Float16* __restrict__ W2T,
    float* __restrict__ H2) {
  __shared__ _Float16 As[16 * HDIM];             // 8 KB A tile (16 x 256)
  const int rowbase = blockIdx.x * 16;
  const int tid = threadIdx.x;

#if HAVE_TDM
  if (tid < 32) {
    unsigned long long gaddr =
        (unsigned long long)(uintptr_t)(S + (size_t)rowbase * HDIM);
    unsigned lds_off = (unsigned)(uintptr_t)&As[0];   // flat[31:0] == LDS byte addr
    u32x4 g0;
    g0[0] = 1u;                                  // count=1, user descriptor
    g0[1] = lds_off;                             // lds_addr
    g0[2] = (unsigned)(gaddr & 0xFFFFFFFFu);     // global_addr[31:0]
    g0[3] = (unsigned)((gaddr >> 32) & 0x1FFFFFFu) | 0x80000000u; // addr[56:32]|type=2
    i32x8 g1;
    g1[0] = 0x00010000;                          // data_size=1 (2B), no multicast
    g1[1] = (int)((unsigned)HDIM << 16);         // tensor_dim0[15:0]=256
    g1[2] = (int)((unsigned)(N_NODES & 0xFFFF) << 16); // dim0 hi=0 | tensor_dim1 lo
    g1[3] = (int)((unsigned)HDIM << 16) | (N_NODES >> 16); // dim1 hi | tile_dim0=256
    g1[4] = 16;                                  // tile_dim1=16, tile_dim2=0
    g1[5] = HDIM;                                // tensor_dim0_stride=256
    g1[6] = 0;
    g1[7] = 0;
    i32x4 z4 = {0, 0, 0, 0};
#if __clang_major__ >= 23
    i32x8 z8 = {0, 0, 0, 0, 0, 0, 0, 0};
    __builtin_amdgcn_tensor_load_to_lds(g0, g1, z4, z4, z8, 0);
#else
    __builtin_amdgcn_tensor_load_to_lds(g0, g1, z4, z4, 0);
#endif
    __builtin_amdgcn_s_wait_tensorcnt(0);
  }
  __syncthreads();
#else
  // cooperative fallback: 256 thr * 16 halves
  {
    const v8h* sp = (const v8h*)(S + (size_t)rowbase * HDIM);
    v8h* dp = (v8h*)&As[0];
#pragma unroll
    for (int j = 0; j < 2; ++j) dp[tid + j * 256] = sp[tid + j * 256];
  }
  __syncthreads();
#endif

  const int lane = tid & 31, wave = tid >> 5;
  const int m    = lane & 15;
  const int ksel = (lane & 16) ? 8 : 0;
  const int moff = (lane & 16) ? 8 : 0;

#pragma unroll
  for (int ct = 0; ct < 2; ++ct) {
    const int colbase = (wave * 2 + ct) * 16;
    v8f c = {};
#pragma unroll
    for (int kb = 0; kb < HDIM / 32; ++kb) {
      v16h a = load_frag(&As[m * HDIM + kb * 32 + ksel]);
      v16h b = load_frag(&W2T[(size_t)(colbase + m) * HDIM + kb * 32 + ksel]);
      c = __builtin_amdgcn_wmma_f32_16x16x32_f16(false, a, false, b,
                                                 (short)0, c, false, false);
    }
    const int n = colbase + m;
#pragma unroll
    for (int r = 0; r < 8; ++r)
      H2[(size_t)(rowbase + r + moff) * HDIM + n] = c[r];
  }
}

// --------------------- edge scatter: mean aggregation ----------------------
// one wave32 per edge; lane handles 8 features; src/dst forced to SGPRs.
__global__ __launch_bounds__(256) void scatter_mean_kernel(
    const float* __restrict__ H1, const int* __restrict__ src,
    const int* __restrict__ dst, float* __restrict__ agg) {
  long long tid = (long long)blockIdx.x * blockDim.x + threadIdx.x;
  int e = __builtin_amdgcn_readfirstlane((int)(tid >> 5));
  if (e >= N_EDGES) return;
  int f = (threadIdx.x & 31) * 8;
  int s = src[e], d = dst[e];                    // scalar (s_load) candidates
  const float* gp = H1 + (size_t)s * HDIM + f;
  __builtin_prefetch(gp + HDIM, 0, 0);           // global_prefetch_b8
  float4 v0 = *(const float4*)gp;
  float4 v1 = *(const float4*)(gp + 4);
  float* op = agg + (size_t)d * HDIM + f;
  atomAddF(op + 0, v0.x); atomAddF(op + 1, v0.y);
  atomAddF(op + 2, v0.z); atomAddF(op + 3, v0.w);
  atomAddF(op + 4, v1.x); atomAddF(op + 5, v1.y);
  atomAddF(op + 6, v1.z); atomAddF(op + 7, v1.w);
}

// -------------------- edge scatter: GCN-normalized agg ---------------------
__global__ __launch_bounds__(256) void scatter_gcn_kernel(
    const float* __restrict__ H2, const int* __restrict__ src,
    const int* __restrict__ dst, const float* __restrict__ ew,
    float* __restrict__ agg) {
  long long tid = (long long)blockIdx.x * blockDim.x + threadIdx.x;
  int e = __builtin_amdgcn_readfirstlane((int)(tid >> 5));
  if (e >= N_EDGES) return;
  int f = (threadIdx.x & 31) * 8;
  int s = src[e], d = dst[e];
  float w = ew[e];
  const float* gp = H2 + (size_t)s * HDIM + f;
  __builtin_prefetch(gp + HDIM, 0, 0);
  float4 v0 = *(const float4*)gp;
  float4 v1 = *(const float4*)(gp + 4);
  float* op = agg + (size_t)d * HDIM + f;
  atomAddF(op + 0, v0.x * w); atomAddF(op + 1, v0.y * w);
  atomAddF(op + 2, v0.z * w); atomAddF(op + 3, v0.w * w);
  atomAddF(op + 4, v1.x * w); atomAddF(op + 5, v1.y * w);
  atomAddF(op + 6, v1.z * w); atomAddF(op + 7, v1.w * w);
}

// ------------------------------- LIF update --------------------------------
// folds the mean divide; emits binary spikes as packed f16 for GEMM2.
__global__ __launch_bounds__(256) void lif_kernel(
    const float* __restrict__ agg, const float* __restrict__ deg,
    float* __restrict__ vbuf, _Float16* __restrict__ sbuf) {
  long long i4 = (long long)blockIdx.x * blockDim.x + threadIdx.x;
  if (i4 >= (long long)N_NODES * HDIM / 4) return;
  int row = __builtin_amdgcn_readfirstlane((int)(i4 >> 6)); // wave-uniform
  float invd = 1.0f / deg[row];
  float4 a = ((const float4*)agg)[i4];
  float4 v = ((const float4*)vbuf)[i4];
  h4 sp;
  float s0, s1, s2, s3;
  v.x += (a.x * invd - v.x) * INV_TAU; s0 = (v.x >= THRV) ? 1.f : 0.f; v.x *= (1.f - s0);
  v.y += (a.y * invd - v.y) * INV_TAU; s1 = (v.y >= THRV) ? 1.f : 0.f; v.y *= (1.f - s1);
  v.z += (a.z * invd - v.z) * INV_TAU; s2 = (v.z >= THRV) ? 1.f : 0.f; v.z *= (1.f - s2);
  v.w += (a.w * invd - v.w) * INV_TAU; s3 = (v.w >= THRV) ? 1.f : 0.f; v.w *= (1.f - s3);
  ((float4*)vbuf)[i4] = v;
  sp[0] = (_Float16)s0; sp[1] = (_Float16)s1; sp[2] = (_Float16)s2; sp[3] = (_Float16)s3;
  *(h4*)(sbuf + i4 * 4) = sp;
}

// ------------------- pooling (+ GCN self-loop term folded) -----------------
__global__ __launch_bounds__(256) void pool_kernel(
    const float* __restrict__ agg2, const float* __restrict__ H2,
    const float* __restrict__ isw, const int* __restrict__ batch,
    float* __restrict__ pool) {
  long long i4 = (long long)blockIdx.x * blockDim.x + threadIdx.x;
  if (i4 >= (long long)N_NODES * HDIM / 4) return;
  int node = __builtin_amdgcn_readfirstlane((int)(i4 >> 6)); // wave-uniform
  int f = ((int)i4 & 63) * 4;
  float iw = isw[node];
  float w2 = iw * iw;                            // self-loop weight
  int g = batch[node];
  float4 a = ((const float4*)agg2)[i4];
  float4 h = ((const float4*)H2)[i4];
  float* pp = pool + (size_t)g * HDIM + f;
  atomAddF(pp + 0, a.x + h.x * w2);
  atomAddF(pp + 1, a.y + h.y * w2);
  atomAddF(pp + 2, a.z + h.z * w2);
  atomAddF(pp + 3, a.w + h.w * w2);
}

// --------------------- final readout (bias algebra folded) -----------------
// out[g,o] = sum_h (pool[g,h]/(cnt*T) + b2[h]) * Wlin[h,o] + blin[o]
__global__ __launch_bounds__(256) void readout_kernel(
    const float* __restrict__ pool, const float* __restrict__ cnt,
    const float* __restrict__ b2, const float* __restrict__ Wlin,
    const float* __restrict__ blin, float* __restrict__ out) {
  int idx = blockIdx.x * blockDim.x + threadIdx.x;
  if (idx >= NGRAPH * OUTC) return;
  int g = idx / OUTC, o = idx % OUTC;
  float invcT = 1.0f / (fmaxf(cnt[g], 1.0f) * (float)TSTEPS);
  float sum = blin[o];
  const float* pg = pool + (size_t)g * HDIM;
#pragma unroll 8
  for (int h = 0; h < HDIM; ++h)
    sum += (pg[h] * invcT + b2[h]) * Wlin[h * OUTC + o];
  out[idx] = sum;
}

// ---------------------------------------------------------------------------
extern "C" void kernel_launch(void* const* d_in, const int* in_sizes, int n_in,
                              void* d_out, int out_size, void* d_ws, size_t ws_size,
                              hipStream_t stream) {
  const float* x     = (const float*)d_in[0];
  const float* W1    = (const float*)d_in[1];
  const float* W2    = (const float*)d_in[2];
  const float* b2    = (const float*)d_in[3];
  const float* Wlin  = (const float*)d_in[4];
  const float* blin  = (const float*)d_in[5];
  const int*   src   = (const int*)d_in[6];
  const int*   dst   = (const int*)d_in[7];
  const int*   batch = (const int*)d_in[8];
  float* out = (float*)d_out;

  // ---- workspace carve-out (256B aligned) ----
  char* w = (char*)d_ws;
  size_t off = 0;
  auto carve = [&](size_t bytes) {
    char* p = w + off;
    off = (off + bytes + 255) & ~(size_t)255;
    return p;
  };
  const size_t NH = (size_t)N_NODES * HDIM;
  float*    bufA = (float*)carve(NH * 4);             // H1 / H2 (reused)
  float*    bufB = (float*)carve(NH * 4);             // agg1 / agg2 (reused)
  float*    vbuf = (float*)carve(NH * 4);             // LIF membrane
  _Float16* sbuf = (_Float16*)carve(NH * 2);          // binary spikes f16
  float*    pool = (float*)carve((size_t)NGRAPH * HDIM * 4);
  float*    degc = (float*)carve((size_t)N_NODES * 4);
  float*    isw  = (float*)carve((size_t)N_NODES * 4);
  float*    cnt  = (float*)carve((size_t)NGRAPH * 4);
  float*    ew   = (float*)carve((size_t)N_EDGES * 4);
  _Float16* W1T  = (_Float16*)carve((size_t)F_INP * HDIM * 2);
  _Float16* W2T  = (_Float16*)carve((size_t)HDIM * HDIM * 2);

  const int TB = 256;
  auto blocks  = [](long long n, int tb) { return (int)((n + tb - 1) / tb); };

  // ---- one-time precompute ----
  zero4_kernel<<<blocks(NH / 4, TB), TB, 0, stream>>>((float4*)vbuf, NH / 4);
  zero4_kernel<<<blocks((size_t)NGRAPH * HDIM / 4, TB), TB, 0, stream>>>(
      (float4*)pool, (size_t)NGRAPH * HDIM / 4);
  zero4_kernel<<<blocks(N_NODES / 4, TB), TB, 0, stream>>>((float4*)degc, N_NODES / 4);
  zero4_kernel<<<blocks(NGRAPH / 4, TB), TB, 0, stream>>>((float4*)cnt, NGRAPH / 4);

  degree_kernel<<<blocks(N_EDGES, TB), TB, 0, stream>>>(dst, degc);
  graphcnt_kernel<<<blocks(N_NODES, TB), TB, 0, stream>>>(batch, cnt);
  finalize_deg_kernel<<<blocks(N_NODES, TB), TB, 0, stream>>>(degc, isw);
  edgew_kernel<<<blocks(N_EDGES, TB), TB, 0, stream>>>(src, dst, isw, ew);
  convert_w1_kernel<<<blocks(F_INP * HDIM, TB), TB, 0, stream>>>(W1, W1T);
  convert_w2_kernel<<<blocks(HDIM * HDIM, TB), TB, 0, stream>>>(W2, W2T);

  const int rowTiles  = N_NODES / 16;                  // 3125
  const long long eth = (long long)N_EDGES * 32;       // 1 wave per edge
  const long long nh4 = NH / 4;

  // ---- temporal loop ----
  for (int t = 0; t < TSTEPS; ++t) {
    zero4_kernel<<<blocks(nh4, TB), TB, 0, stream>>>((float4*)bufB, nh4);
    spike_gemm1_kernel<<<rowTiles, TB, 0, stream>>>(x, W1T, bufA, t);
    scatter_mean_kernel<<<blocks(eth, TB), TB, 0, stream>>>(bufA, src, dst, bufB);
    lif_kernel<<<blocks(nh4, TB), TB, 0, stream>>>(bufB, degc, vbuf, sbuf);
    spike_gemm2_kernel<<<rowTiles, TB, 0, stream>>>(sbuf, W2T, bufA);
    zero4_kernel<<<blocks(nh4, TB), TB, 0, stream>>>((float4*)bufB, nh4);
    scatter_gcn_kernel<<<blocks(eth, TB), TB, 0, stream>>>(bufA, src, dst, ew, bufB);
    pool_kernel<<<blocks(nh4, TB), TB, 0, stream>>>(bufB, bufA, isw, batch, pool);
  }

  // ---- one-shot readout with folded biases ----
  readout_kernel<<<blocks(NGRAPH * OUTC, TB), TB, 0, stream>>>(
      pool, cnt, b2, Wlin, blin, out);
}
